// MambaLayer_8572754722955
// MI455X (gfx1250) — compile-verified, matrix operand order checked by reading, not run on hardware
//
#include <hip/hip_runtime.h>
#include <hip/hip_bf16.h>

typedef __attribute__((ext_vector_type(16))) _Float16 v16h;
typedef __attribute__((ext_vector_type(8)))  _Float16 v8h;
typedef __attribute__((ext_vector_type(8)))  float    v8f;

#define BB  4
#define DM  384
#define LL  4096
#define TT  (BB*LL)     /* 16384 tokens */
#define DI  768
#define DS  16
#define DTR 24
#define NC  16          /* scan chunks */
#define CL  (LL/NC)     /* 256 steps per chunk */

// ---------------- weight conversion (fp32 -> f16, with zero padding) ----------
__global__ void cvt_plain(const float* __restrict__ in, _Float16* __restrict__ out, int n) {
    int i = blockIdx.x * blockDim.x + threadIdx.x;
    if (i < n) out[i] = (_Float16)in[i];
}
__global__ void cvt_pad_rows(const float* __restrict__ in, _Float16* __restrict__ out,
                             int rin, int rout, int cols) {
    int i = blockIdx.x * blockDim.x + threadIdx.x;
    if (i < rout * cols) {
        int r = i / cols;
        out[i] = (r < rin) ? (_Float16)in[i] : (_Float16)0.0f;
    }
}
__global__ void cvt_pad_cols(const float* __restrict__ in, _Float16* __restrict__ out,
                             int rows, int cin, int cout) {
    int i = blockIdx.x * blockDim.x + threadIdx.x;
    if (i < rows * cout) {
        int r = i / cout, c = i % cout;
        out[i] = (c < cin) ? (_Float16)in[r * cin + c] : (_Float16)0.0f;
    }
}

// ---------------- LayerNorm over channels: one wave32 per token ---------------
__global__ void ln_kernel(const float* __restrict__ x, const float* __restrict__ w,
                          const float* __restrict__ b, _Float16* __restrict__ xn) {
    int wave = (blockIdx.x * blockDim.x + threadIdx.x) >> 5;
    int lane = threadIdx.x & 31;
    if (wave >= TT) return;
    int bidx = wave / LL, l = wave % LL;
    const float* xb = x + (size_t)bidx * DM * LL + l;   // x[b, d, l], stride L per d
    float v[DM / 32];
    float s = 0.f;
#pragma unroll
    for (int i = 0; i < DM / 32; ++i) { v[i] = xb[(size_t)(lane + 32 * i) * LL]; s += v[i]; }
#pragma unroll
    for (int o = 16; o > 0; o >>= 1) s += __shfl_xor(s, o, 32);
    float mu = s * (1.0f / DM);
    float s2 = 0.f;
#pragma unroll
    for (int i = 0; i < DM / 32; ++i) { float d = v[i] - mu; s2 += d * d; }
#pragma unroll
    for (int o = 16; o > 0; o >>= 1) s2 += __shfl_xor(s2, o, 32);
    float rs = rsqrtf(s2 * (1.0f / DM) + 1e-5f);
#pragma unroll
    for (int i = 0; i < DM / 32; ++i) {
        int d = lane + 32 * i;
        xn[(size_t)wave * DM + d] = (_Float16)((v[i] - mu) * rs * w[d] + b[d]);
    }
}

// ------- WMMA GEMM, 16x64 tile per wave: C[M,N] = A[M,K] * W[N,K]^T ----------
// EPI 0: plain f32 store   EPI 1: softplus(acc + bias[n])
// EPI 2: transposed residual store into (b, n, l) layout
template <int EPI>
__global__ void gemm_wmma(const _Float16* __restrict__ A, const _Float16* __restrict__ Bw,
                          float* __restrict__ C, int M, int N, int K, int ldc,
                          const float* __restrict__ bias,
                          const float* __restrict__ resid, float* __restrict__ outp) {
    int tiles_n = N >> 6;                       // groups of 4 16-wide n-tiles
    int wave = (blockIdx.x * blockDim.x + threadIdx.x) >> 5;
    int total = (M >> 4) * tiles_n;
    if (wave >= total) return;                  // wave-uniform: EXEC stays all-ones
    int tm = wave / tiles_n, tg = wave % tiles_n;
    int lane = threadIdx.x & 31;
    int mr = lane & 15, hi = lane >> 4;
    // CDNA5 16-bit A/B lane layout: lane (hi,m) holds K = hi*8.. and 16+hi*8.. per 32-chunk
    const _Float16* Ap = A  + (size_t)(tm * 16 + mr) * K + hi * 8;
    const _Float16* Bp = Bw + (size_t)(tg * 64 + mr) * K + hi * 8;
    v8f acc0 = {}, acc1 = {}, acc2 = {}, acc3 = {};
    for (int kb = 0; kb < K; kb += 32) {
        union { v16h v; v8h h[2]; } a, b0, b1, b2, b3;
        a.h[0]  = *(const v8h*)(Ap + kb);
        a.h[1]  = *(const v8h*)(Ap + kb + 16);
        b0.h[0] = *(const v8h*)(Bp + kb);              b0.h[1] = *(const v8h*)(Bp + kb + 16);
        b1.h[0] = *(const v8h*)(Bp + 16 * K + kb);     b1.h[1] = *(const v8h*)(Bp + 16 * K + kb + 16);
        b2.h[0] = *(const v8h*)(Bp + 32 * K + kb);     b2.h[1] = *(const v8h*)(Bp + 32 * K + kb + 16);
        b3.h[0] = *(const v8h*)(Bp + 48 * K + kb);     b3.h[1] = *(const v8h*)(Bp + 48 * K + kb + 16);
        acc0 = __builtin_amdgcn_wmma_f32_16x16x32_f16(false, a.v, false, b0.v, (short)0, acc0, false, false);
        acc1 = __builtin_amdgcn_wmma_f32_16x16x32_f16(false, a.v, false, b1.v, (short)0, acc1, false, false);
        acc2 = __builtin_amdgcn_wmma_f32_16x16x32_f16(false, a.v, false, b2.v, (short)0, acc2, false, false);
        acc3 = __builtin_amdgcn_wmma_f32_16x16x32_f16(false, a.v, false, b3.v, (short)0, acc3, false, false);
    }
#pragma unroll
    for (int j = 0; j < 4; ++j) {
        v8f acc = (j == 0) ? acc0 : (j == 1) ? acc1 : (j == 2) ? acc2 : acc3;
        int col = tg * 64 + j * 16 + mr;
#pragma unroll
        for (int r = 0; r < 8; ++r) {
            int row = tm * 16 + r + hi * 8;     // C layout: VGPR r -> M = r (+8 for hi half)
            float v = acc[r];
            if (EPI == 0) {
                C[(size_t)row * ldc + col] = v;
            } else if (EPI == 1) {
                float t = v + bias[col];
                C[(size_t)row * ldc + col] = (t > 20.f) ? t : log1pf(expf(t));
            } else {
                int bb = row >> 12;             // token / 4096
                int l  = row & 4095;
                size_t o = ((size_t)(bb * DM + col)) * LL + l;
                outp[o] = v + resid[o];
            }
        }
    }
}

// ---------------- depthwise causal conv1d (k=4) + SiLU ------------------------
__global__ void conv_silu(const float* __restrict__ xz, const float* __restrict__ cw,
                          const float* __restrict__ cb,
                          float* __restrict__ xaf, _Float16* __restrict__ xah) {
    int i = blockIdx.x * blockDim.x + threadIdx.x;
    if (i >= TT * DI) return;
    int t = i / DI, c = i % DI;
    int b = t >> 12, l = t & 4095;
    float acc = cb[c];
#pragma unroll
    for (int j = 0; j < 4; ++j) {
        int ll = l - 3 + j;
        if (ll >= 0) acc += cw[c * 4 + j] * xz[((size_t)(b * LL + ll)) * (2 * DI) + c];
    }
    float s = acc / (1.f + expf(-acc));
    xaf[i] = s;
    xah[i] = (_Float16)s;
}

// ---------------- split x_dbl(pad64) into dt(pad32,f16), Bm, Cm ---------------
__global__ void split_kernel(const float* __restrict__ xdbl, _Float16* __restrict__ dtp,
                             float* __restrict__ Bm, float* __restrict__ Cm) {
    int i = blockIdx.x * blockDim.x + threadIdx.x;
    if (i >= TT * 64) return;
    int t = i >> 6, j = i & 63;
    float v = xdbl[i];
    if (j < 32) dtp[t * 32 + j] = (j < DTR) ? (_Float16)v : (_Float16)0.0f;
    if (j >= 24 && j < 40)       Bm[t * DS + (j - 24)] = v;
    else if (j >= 40 && j < 56)  Cm[t * DS + (j - 40)] = v;
}

// -------- chunked selective scan, pass 1: per-chunk summaries -----------------
// blockIdx = pair*NC + chunk; wave handles 2 (b,ch); lane = state index n
__global__ void scan_pass1(const float* __restrict__ delta, const float* __restrict__ u,
                           const float* __restrict__ Bm, const float* __restrict__ A_log,
                           float* __restrict__ ap_out, float* __restrict__ he_out) {
    int lane = threadIdx.x & 31;
    int n = lane & 15, sel = lane >> 4;
    int chunk = blockIdx.x % NC, pair = blockIdx.x / NC;
    int ch = pair * 2 + sel;                   // 0 .. B*DI-1
    int b = ch / DI, c = ch % DI;
    float a = -expf(A_log[c * DS + n]);
    float h = 0.f, ap = 1.f;
    int l0 = chunk * CL;
    for (int l = l0; l < l0 + CL; ++l) {
        int t = b * LL + l;
        __builtin_prefetch(&delta[(size_t)(t + 16) * DI + c], 0, 0);
        __builtin_prefetch(&Bm[(t + 16) * DS + n], 0, 0);
        float dv = delta[(size_t)t * DI + c];
        float uv = u[(size_t)t * DI + c];
        float bn = Bm[t * DS + n];
        float da = __expf(dv * a);
        ap *= da;
        h = h * da + dv * uv * bn;
    }
    int o = (ch * NC + chunk) * DS + n;
    ap_out[o] = ap;
    he_out[o] = h;
}

// -------- pass 2: prefix-compose chunk start states (tiny, sequential in NC) --
__global__ void scan_combine(const float* __restrict__ ap, const float* __restrict__ he,
                             float* __restrict__ g) {
    int i = blockIdx.x * blockDim.x + threadIdx.x;     // over (B*DI)*DS
    if (i >= BB * DI * DS) return;
    int ch = i / DS, n = i % DS;
    float hg = 0.f;
#pragma unroll
    for (int k = 0; k < NC; ++k) {
        int o = (ch * NC + k) * DS + n;
        g[o] = hg;
        hg = hg * ap[o] + he[o];
    }
}

// -------- pass 3: re-run chunk from correct start state, emit gated y ---------
__global__ void scan_pass2(const float* __restrict__ delta, const float* __restrict__ u,
                           const float* __restrict__ Bm, const float* __restrict__ Cm,
                           const float* __restrict__ A_log, const float* __restrict__ Dp,
                           const float* __restrict__ xz, const float* __restrict__ g,
                           _Float16* __restrict__ yh) {
    int lane = threadIdx.x & 31;
    int n = lane & 15, sel = lane >> 4;
    int chunk = blockIdx.x % NC, pair = blockIdx.x / NC;
    int ch = pair * 2 + sel;
    int b = ch / DI, c = ch % DI;
    float a = -expf(A_log[c * DS + n]);
    float dpc = Dp[c];
    float h = g[(ch * NC + chunk) * DS + n];
    int l0 = chunk * CL;
    for (int l = l0; l < l0 + CL; ++l) {
        int t = b * LL + l;
        __builtin_prefetch(&delta[(size_t)(t + 16) * DI + c], 0, 0);
        __builtin_prefetch(&Bm[(t + 16) * DS + n], 0, 0);
        __builtin_prefetch(&Cm[(t + 16) * DS + n], 0, 0);
        float dv = delta[(size_t)t * DI + c];
        float uv = u[(size_t)t * DI + c];
        float bn = Bm[t * DS + n];
        float cn = Cm[t * DS + n];
        h = h * __expf(dv * a) + dv * uv * bn;
        float y = h * cn;
        y += __shfl_xor(y, 1, 32);
        y += __shfl_xor(y, 2, 32);
        y += __shfl_xor(y, 4, 32);
        y += __shfl_xor(y, 8, 32);
        if (n == 0) {
            float yt = y + uv * dpc;
            float z = xz[(size_t)t * (2 * DI) + DI + c];
            float gz = z / (1.f + expf(-z));   // silu(z)
            yh[(size_t)t * DI + c] = (_Float16)(yt * gz);
        }
    }
}

extern "C" void kernel_launch(void* const* d_in, const int* in_sizes, int n_in,
                              void* d_out, int out_size, void* d_ws, size_t ws_size,
                              hipStream_t stream) {
    const float* x      = (const float*)d_in[0];
    const float* ln_w   = (const float*)d_in[1];
    const float* ln_b   = (const float*)d_in[2];
    const float* W_in   = (const float*)d_in[3];   // (1536, 384)
    const float* conv_w = (const float*)d_in[4];   // (768, 1, 4)
    const float* conv_b = (const float*)d_in[5];   // (768)
    const float* W_xp   = (const float*)d_in[6];   // (56, 768)
    const float* W_dt   = (const float*)d_in[7];   // (768, 24)
    const float* b_dt   = (const float*)d_in[8];   // (768)
    const float* A_log  = (const float*)d_in[9];   // (768, 16)
    const float* Dp     = (const float*)d_in[10];  // (768)
    const float* W_out  = (const float*)d_in[11];  // (384, 768)
    float* out = (float*)d_out;

    char* p = (char*)d_ws;
    _Float16* xn_h   = (_Float16*)p; p += (size_t)TT * DM * 2;
    _Float16* Win_h  = (_Float16*)p; p += (size_t)1536 * DM * 2;
    _Float16* Wxp_h  = (_Float16*)p; p += (size_t)64 * DI * 2;      // padded 56->64 rows
    _Float16* Wdt_h  = (_Float16*)p; p += (size_t)DI * 32 * 2;      // padded 24->32 cols
    _Float16* Wout_h = (_Float16*)p; p += (size_t)DM * DI * 2;
    float*    xz     = (float*)p;    p += (size_t)TT * 1536 * 4;
    float*    xa_f   = (float*)p;    p += (size_t)TT * DI * 4;
    _Float16* xa_h   = (_Float16*)p; p += (size_t)TT * DI * 2;
    float*    xdbl   = (float*)p;    p += (size_t)TT * 64 * 4;      // stride-64 padded
    _Float16* dt_h   = (_Float16*)p; p += (size_t)TT * 32 * 2;
    float*    Bmat   = (float*)p;    p += (size_t)TT * DS * 4;
    float*    Cmat   = (float*)p;    p += (size_t)TT * DS * 4;
    float*    dlt    = (float*)p;    p += (size_t)TT * DI * 4;
    _Float16* y_h    = (_Float16*)p; p += (size_t)TT * DI * 2;
    float*    s_ap   = (float*)p;    p += (size_t)BB * DI * DS * NC * 4;
    float*    s_he   = (float*)p;    p += (size_t)BB * DI * DS * NC * 4;
    float*    s_g    = (float*)p;    p += (size_t)BB * DI * DS * NC * 4;

    // weight conversions
    cvt_plain<<<(1536 * DM + 255) / 256, 256, 0, stream>>>(W_in, Win_h, 1536 * DM);
    cvt_pad_rows<<<(64 * DI + 255) / 256, 256, 0, stream>>>(W_xp, Wxp_h, 56, 64, DI);
    cvt_pad_cols<<<(DI * 32 + 255) / 256, 256, 0, stream>>>(W_dt, Wdt_h, DI, DTR, 32);
    cvt_plain<<<(DM * DI + 255) / 256, 256, 0, stream>>>(W_out, Wout_h, DM * DI);

    // LayerNorm
    ln_kernel<<<TT / 8, 256, 0, stream>>>(x, ln_w, ln_b, xn_h);

    // in_proj: (16384 x 384) * (384 x 1536)
    {
        int tiles = (TT / 16) * (1536 / 64);
        gemm_wmma<0><<<(tiles + 7) / 8, 256, 0, stream>>>(xn_h, Win_h, xz, TT, 1536, DM,
                                                          1536, nullptr, nullptr, nullptr);
    }
    // depthwise conv + SiLU
    conv_silu<<<(TT * DI + 255) / 256, 256, 0, stream>>>(xz, conv_w, conv_b, xa_f, xa_h);

    // x_proj: (16384 x 768) * (768 x 64pad)
    {
        int tiles = (TT / 16) * (64 / 64);
        gemm_wmma<0><<<(tiles + 7) / 8, 256, 0, stream>>>(xa_h, Wxp_h, xdbl, TT, 64, DI,
                                                          64, nullptr, nullptr, nullptr);
    }
    split_kernel<<<(TT * 64 + 255) / 256, 256, 0, stream>>>(xdbl, dt_h, Bmat, Cmat);

    // dt_proj + softplus: (16384 x 32pad) * (32 x 768)
    {
        int tiles = (TT / 16) * (DI / 64);
        gemm_wmma<1><<<(tiles + 7) / 8, 256, 0, stream>>>(dt_h, Wdt_h, dlt, TT, DI, 32,
                                                          DI, b_dt, nullptr, nullptr);
    }
    // chunk-parallel selective scan (pass1 summaries -> prefix -> pass2 emit)
    scan_pass1<<<(BB * DI / 2) * NC, 32, 0, stream>>>(dlt, xa_f, Bmat, A_log, s_ap, s_he);
    scan_combine<<<(BB * DI * DS + 255) / 256, 256, 0, stream>>>(s_ap, s_he, s_g);
    scan_pass2<<<(BB * DI / 2) * NC, 32, 0, stream>>>(dlt, xa_f, Bmat, Cmat, A_log, Dp,
                                                      xz, s_g, y_h);

    // out_proj + residual (transposed store): (16384 x 768) * (768 x 384)
    {
        int tiles = (TT / 16) * (DM / 64);
        gemm_wmma<2><<<(tiles + 7) / 8, 256, 0, stream>>>(y_h, Wout_h, nullptr, TT, DM, DI,
                                                          0, nullptr, x, out);
    }
}
